// TernaryLinear_28080496181413
// MI455X (gfx1250) — compile-verified, matrix operand order checked by reading, not run on hardware
//
#include <hip/hip_runtime.h>

// ---- types matching the probe-verified gfx1250 WMMA builtins ----
typedef __attribute__((ext_vector_type(16))) __bf16 v16bf;
typedef __attribute__((ext_vector_type(8)))  float  v8f;

// Problem dims (fixed by the reference)
constexpr int M_ = 8192;
constexpr int N_ = 4096;
constexpr int K_ = 4096;

// GEMM tiling (fast path)
constexpr int BM  = 128;          // block tile M
constexpr int BN  = 128;          // block tile N
constexpr int BK2 = 64;           // K step = 2x WMMA bf16 K-depth
constexpr int KSTR = BK2 + 8;     // padded LDS row stride (bf16), 144B = 16B-aligned

__device__ __forceinline__ void split_bf16(float v, __bf16& h, __bf16& l) {
    h = (__bf16)v;                 // RNE to bf16 (hi part)
    l = (__bf16)(v - (float)h);    // residual (lo part); hi+lo ~16 mantissa bits
}

__device__ __forceinline__ __bf16 tern_bf16(float v) {
    float r = rintf(v);            // round half-to-even, matches jnp.round
    float t = (r > 0.f) ? 1.f : ((r < 0.f) ? -1.f : 0.f);
    return (__bf16)t;              // {-1,0,1} exact in bf16
}

// CDNA5 async global->LDS copy, 16B per lane, tracked by ASYNCcnt.
// ldsAddr: wave-relative LDS byte address (low 32 bits of generic LDS pointer).
__device__ __forceinline__ void async_copy_b128(unsigned ldsAddr, const void* gptr) {
    asm volatile("global_load_async_to_lds_b128 %0, %1, off"
                 :: "v"(ldsAddr), "v"(gptr)
                 : "memory");
}

__device__ __forceinline__ void wait_asynccnt0() {
    asm volatile("s_wait_asynccnt 0x0" ::: "memory");
}

// ---------------- preconversion kernels (streaming, memory-bound) ----------------

__global__ __launch_bounds__(256)
void ternarize_w_kernel(const float* __restrict__ w, __bf16* __restrict__ wt, int n4) {
    int i = blockIdx.x * blockDim.x + threadIdx.x;
    const int stride = gridDim.x * blockDim.x;
    for (; i < n4; i += stride) {
        const float4 v = ((const float4*)w)[i];
        union { __bf16 b[4]; uint2 u; } p;
        p.b[0] = tern_bf16(v.x); p.b[1] = tern_bf16(v.y);
        p.b[2] = tern_bf16(v.z); p.b[3] = tern_bf16(v.w);
        *(uint2*)(wt + (size_t)i * 4) = p.u;
    }
}

__global__ __launch_bounds__(256)
void split_x_kernel(const float* __restrict__ x,
                    __bf16* __restrict__ xhi, __bf16* __restrict__ xlo, int n4) {
    int i = blockIdx.x * blockDim.x + threadIdx.x;
    const int stride = gridDim.x * blockDim.x;
    for (; i < n4; i += stride) {
        const float4 v = ((const float4*)x)[i];
        union { __bf16 b[4]; uint2 u; } h, l;
        split_bf16(v.x, h.b[0], l.b[0]);
        split_bf16(v.y, h.b[1], l.b[1]);
        split_bf16(v.z, h.b[2], l.b[2]);
        split_bf16(v.w, h.b[3], l.b[3]);
        *(uint2*)(xhi + (size_t)i * 4) = h.u;
        *(uint2*)(xlo + (size_t)i * 4) = l.u;
    }
}

// ---------------- fast GEMM: bf16 WMMA + async global->LDS staging ----------------

__global__ __launch_bounds__(256)
void ternary_gemm_bf16_fast(const __bf16* __restrict__ xhi,
                            const __bf16* __restrict__ xlo,
                            const __bf16* __restrict__ wt,
                            float* __restrict__ out) {
    __shared__ __bf16 sXhi[BM * KSTR];
    __shared__ __bf16 sXlo[BM * KSTR];
    __shared__ __bf16 sW  [BN * KSTR];

    const int tid  = threadIdx.x;
    const int lane = tid & 31;
    const int waveId = tid >> 5;           // 0..7
    const int waveM  = waveId & 3;         // 4 waves along M (32 rows each)
    const int waveN  = waveId >> 2;        // 2 waves along N (64 cols each)

    const int mBase = blockIdx.y * BM;
    const int nBase = blockIdx.x * BN;

    // fragment lane decomposition (16-bit A/B layout, wave32)
    const int lr      = lane & 15;
    const int halfSel = lane >> 4;
    const int kOff    = halfSel * 8;

    // global->LDS mapping: 128 rows x 64 bf16 = 1024 16B chunks; 256 thr x 4 passes
    const int chunk   = tid & 7;           // 8 chunks of 8 bf16 per row
    const int rowBase = tid >> 3;          // 0..31

    v8f acc[2][4] = {};

    const int ksteps = K_ / BK2;           // 64
    for (int kt = 0; kt < ksteps; ++kt) {
        const int k0 = kt * BK2;

        if (kt) __syncthreads();           // previous compute done reading LDS

        // ---- stage bf16 tiles: async global->LDS b128 (no VGPR round-trip) ----
#pragma unroll
        for (int p = 0; p < 4; ++p) {
            const int row = rowBase + p * 32;
            const size_t gx = (size_t)(mBase + row) * K_ + k0 + chunk * 8;
            const size_t gw = (size_t)(nBase + row) * K_ + k0 + chunk * 8;
            const int ldsOff = row * KSTR + chunk * 8;
            async_copy_b128((unsigned)(uintptr_t)(sXhi + ldsOff), xhi + gx);
            async_copy_b128((unsigned)(uintptr_t)(sXlo + ldsOff), xlo + gx);
            async_copy_b128((unsigned)(uintptr_t)(sW   + ldsOff), wt  + gw);
        }
        // prefetch next K tile into cache (global_prefetch_b8)
        if (kt + 1 < ksteps) {
            const int kn = k0 + BK2;
            __builtin_prefetch(xhi + (size_t)(mBase + rowBase) * K_ + kn + chunk * 8, 0, 1);
            __builtin_prefetch(wt  + (size_t)(nBase + rowBase) * K_ + kn + chunk * 8, 0, 1);
        }
        wait_asynccnt0();                  // this wave's async copies are in LDS
        __syncthreads();                   // all waves' copies visible

        // ---- 2 K-halves of 32, fragments via ds_load_b128, 32 WMMAs/wave/step ----
        auto loadFrag = [&](const __bf16* base, int row, int kh) -> v16bf {
            const int e0 = row * KSTR + kh + kOff;
            union { uint4 q[2]; v16bf v; } u;
            u.q[0] = *(const uint4*)(base + e0);
            u.q[1] = *(const uint4*)(base + e0 + 16);
            return u.v;
        };

#pragma unroll
        for (int kh = 0; kh < BK2; kh += 32) {
            v16bf aHi[2], aLo[2], bF[4];
#pragma unroll
            for (int mi = 0; mi < 2; ++mi) {
                const int row = waveM * 32 + mi * 16 + lr;
                aHi[mi] = loadFrag(sXhi, row, kh);
                aLo[mi] = loadFrag(sXlo, row, kh);
            }
#pragma unroll
            for (int ni = 0; ni < 4; ++ni) {
                const int col = waveN * 64 + ni * 16 + lr;
                bF[ni] = loadFrag(sW, col, kh);
            }
#pragma unroll
            for (int mi = 0; mi < 2; ++mi) {
#pragma unroll
                for (int ni = 0; ni < 4; ++ni) {
                    acc[mi][ni] = __builtin_amdgcn_wmma_f32_16x16x32_bf16(
                        false, aHi[mi], false, bF[ni], (short)0, acc[mi][ni], false, false);
                    acc[mi][ni] = __builtin_amdgcn_wmma_f32_16x16x32_bf16(
                        false, aLo[mi], false, bF[ni], (short)0, acc[mi][ni], false, false);
                }
            }
        }
    }

    // ---- epilogue: C/D layout (VGPR v -> row v + 8*halfSel, col = lane&15) ----
#pragma unroll
    for (int mi = 0; mi < 2; ++mi) {
#pragma unroll
        for (int ni = 0; ni < 4; ++ni) {
            const int gm = mBase + waveM * 32 + mi * 16 + halfSel * 8;
            const int gn = nBase + waveN * 64 + ni * 16 + lr;
            float* o = out + (size_t)gm * N_ + gn;
#pragma unroll
            for (int v = 0; v < 8; ++v) {
                o[(size_t)v * N_] = acc[mi][ni][v];
            }
        }
    }
}

// ---------------- fallback: inline-convert GEMM (ws-free) ----------------

constexpr int BKF  = 32;
constexpr int XSTR = BKF + 8;

__global__ __launch_bounds__(256)
void ternary_linear_wmma_inline(const float* __restrict__ x,
                                const float* __restrict__ w,
                                float* __restrict__ out) {
    __shared__ __bf16 sXhi[BM * XSTR];
    __shared__ __bf16 sXlo[BM * XSTR];
    __shared__ __bf16 sW  [BN * XSTR];

    const int tid  = threadIdx.x;
    const int lane = tid & 31;
    const int waveId = tid >> 5;
    const int waveM  = waveId & 3;
    const int waveN  = waveId >> 2;

    const int mBase = blockIdx.y * BM;
    const int nBase = blockIdx.x * BN;

    const int lr      = lane & 15;
    const int halfSel = lane >> 4;
    const int kOff    = halfSel * 8;

    const int rowInPass = tid >> 3;
    const int col4      = (tid & 7) * 4;

    v8f acc[2][4] = {};

    const int ksteps = K_ / BKF;
    for (int kt = 0; kt < ksteps; ++kt) {
        const int k0 = kt * BKF;

        float4 xr[4], wr[4];
#pragma unroll
        for (int p = 0; p < 4; ++p) {
            const int r = rowInPass + p * 32;
            xr[p] = *(const float4*)(x + (size_t)(mBase + r) * K_ + k0 + col4);
            wr[p] = *(const float4*)(w + (size_t)(nBase + r) * K_ + k0 + col4);
        }

        if (kt) __syncthreads();

#pragma unroll
        for (int p = 0; p < 4; ++p) {
            const int r = rowInPass + p * 32;
            const float xv[4] = {xr[p].x, xr[p].y, xr[p].z, xr[p].w};
            const float wv[4] = {wr[p].x, wr[p].y, wr[p].z, wr[p].w};
#pragma unroll
            for (int j = 0; j < 4; ++j) {
                __bf16 h, l;
                split_bf16(xv[j], h, l);
                sXhi[r * XSTR + col4 + j] = h;
                sXlo[r * XSTR + col4 + j] = l;
                sW  [r * XSTR + col4 + j] = tern_bf16(wv[j]);
            }
        }
        __syncthreads();

        auto loadFrag = [&](const __bf16* base, int row) -> v16bf {
            const int e0 = row * XSTR + kOff;
            union { uint4 q[2]; v16bf v; } u;
            u.q[0] = *(const uint4*)(base + e0);
            u.q[1] = *(const uint4*)(base + e0 + 16);
            return u.v;
        };

        v16bf aHi[2], aLo[2], bF[4];
#pragma unroll
        for (int mi = 0; mi < 2; ++mi) {
            const int row = waveM * 32 + mi * 16 + lr;
            aHi[mi] = loadFrag(sXhi, row);
            aLo[mi] = loadFrag(sXlo, row);
        }
#pragma unroll
        for (int ni = 0; ni < 4; ++ni) {
            const int col = waveN * 64 + ni * 16 + lr;
            bF[ni] = loadFrag(sW, col);
        }

#pragma unroll
        for (int mi = 0; mi < 2; ++mi) {
#pragma unroll
            for (int ni = 0; ni < 4; ++ni) {
                acc[mi][ni] = __builtin_amdgcn_wmma_f32_16x16x32_bf16(
                    false, aHi[mi], false, bF[ni], (short)0, acc[mi][ni], false, false);
                acc[mi][ni] = __builtin_amdgcn_wmma_f32_16x16x32_bf16(
                    false, aLo[mi], false, bF[ni], (short)0, acc[mi][ni], false, false);
            }
        }
    }

#pragma unroll
    for (int mi = 0; mi < 2; ++mi) {
#pragma unroll
        for (int ni = 0; ni < 4; ++ni) {
            const int gm = mBase + waveM * 32 + mi * 16 + halfSel * 8;
            const int gn = nBase + waveN * 64 + ni * 16 + lr;
            float* o = out + (size_t)gm * N_ + gn;
#pragma unroll
            for (int v = 0; v < 8; ++v) {
                o[(size_t)v * N_] = acc[mi][ni][v];
            }
        }
    }
}

// ---------------- launch ----------------

extern "C" void kernel_launch(void* const* d_in, const int* in_sizes, int n_in,
                              void* d_out, int out_size, void* d_ws, size_t ws_size,
                              hipStream_t stream) {
    const float* x = (const float*)d_in[0];   // [8192, 4096] f32
    const float* w = (const float*)d_in[1];   // [4096, 4096] f32
    float* out = (float*)d_out;               // [8192, 4096] f32

    const size_t needW = (size_t)N_ * K_ * sizeof(__bf16);       // 32 MiB
    const size_t needX = (size_t)M_ * K_ * sizeof(__bf16);       // 64 MiB
    const size_t need  = needW + 2 * needX;                      // 160 MiB

    dim3 grid(N_ / BN, M_ / BM);              // 32 x 64 workgroups
    dim3 block(256);                          // 8 waves (wave32)

    if (ws_size >= need && d_ws != nullptr) {
        __bf16* wt  = (__bf16*)d_ws;
        __bf16* xhi = (__bf16*)((char*)d_ws + needW);
        __bf16* xlo = (__bf16*)((char*)d_ws + needW + needX);

        const int nW4 = (N_ * K_) / 4;        // 4,194,304 float4s
        const int nX4 = (M_ * K_) / 4;        // 8,388,608 float4s
        ternarize_w_kernel<<<4096, 256, 0, stream>>>(w, wt, nW4);
        split_x_kernel<<<8192, 256, 0, stream>>>(x, xhi, xlo, nX4);
        ternary_gemm_bf16_fast<<<grid, block, 0, stream>>>(xhi, xlo, wt, out);
    } else {
        ternary_linear_wmma_inline<<<grid, block, 0, stream>>>(x, w, out);
    }
}
